// VisionMamba_71768903516857
// MI455X (gfx1250) — matrix operand order repaired and checked by reading, not compiled, not run on hardware
//
#include <hip/hip_runtime.h>
#include <hip/hip_bf16.h>
#include <cstdint>
#include <cstddef>

// ---------------- model dims ----------------
#define B_     16
#define L_     64
#define D_     256
#define DIN_   512
#define H_     8
#define DH_    64
#define NST_   64
#define CDIM_  640
#define DPROJ_ 1160
#define DPROJP 1280        // DPROJ_ padded to multiple of 128 (GEMM N-pad)
#define NCLS_  1000
#define ROWS_  (B_ * L_)   // 1024
#define EPS_   1e-6f

typedef unsigned short u16;
typedef __attribute__((ext_vector_type(16))) __bf16 v16bf;
typedef __attribute__((ext_vector_type(16))) unsigned short v16u;
typedef __attribute__((ext_vector_type(8)))  float v8f;

static __device__ __forceinline__ u16 f2bf(float f) {
  unsigned u = __builtin_bit_cast(unsigned, f);
  u += 0x7FFFu + ((u >> 16) & 1u);          // round-to-nearest-even
  return (u16)(u >> 16);
}
static __device__ __forceinline__ float siluf(float x) {
  return x / (1.0f + expf(-x));
}

// ---------------- weight conversion ----------------
__global__ void cvt_bf16_k(const float* __restrict__ in, u16* __restrict__ out, int n) {
  int i = blockIdx.x * blockDim.x + threadIdx.x;
  if (i < n) out[i] = f2bf(in[i]);
}
// batched transpose-convert with column padding:
// in (Bn, R, C) -> out (Bn, Cpad, R); out[d][c][r] = in[d][r][c] (pad rows pre-zeroed)
__global__ void cvtT_bf16_k(const float* __restrict__ in, u16* __restrict__ out,
                            int Bn, int R, int C, int Cpad) {
  int i = blockIdx.x * blockDim.x + threadIdx.x;
  if (i >= Bn * R * C) return;
  int d = i / (R * C), rem = i % (R * C);
  int r = rem / C, c = rem % C;
  out[(size_t)d * Cpad * R + (size_t)c * R + r] = f2bf(in[i]);
}

// ---------------- im2col with snake ordering (bf16 A-matrix) ----------------
__global__ void im2col_k(const float* __restrict__ x, u16* __restrict__ A) {
  int idx = blockIdx.x * blockDim.x + threadIdx.x;
  if (idx >= ROWS_ * 768) return;
  int col = idx % 768, row = idx / 768;
  int b = row >> 6, i = row & 63;
  int r8 = i >> 3, c8 = i & 7;
  int lsn = r8 * 8 + ((r8 & 1) ? (7 - c8) : c8);       // SNAKE[i]
  int gy = lsn >> 3, gx = lsn & 7;
  int ch = col >> 8, py = (col & 255) >> 4, px = col & 15;
  float v = x[(((size_t)b * 3 + ch) * 128 + gy * 16 + py) * 128 + gx * 16 + px];
  A[idx] = f2bf(v);
}

__global__ void bias_pos_k(float* __restrict__ t, const float* __restrict__ pb,
                           const float* __restrict__ pos) {
  int idx = blockIdx.x * blockDim.x + threadIdx.x;
  if (idx >= ROWS_ * D_) return;
  int d = idx & (D_ - 1);
  int row = idx >> 8;
  int i = row & 63;
  int r8 = i >> 3, c8 = i & 7;
  int lsn = r8 * 8 + ((r8 & 1) ? (7 - c8) : c8);
  t[idx] += pb[d] + pos[lsn * D_ + d];
}

// ---------------- generic bf16 WMMA GEMM: C[M,N] (+)= alpha * A[M,K] * Bt^T ----------------
// A  row-major bf16 (M x K).
// Bt row-major bf16 (Npad x K) == B[K,N] pre-transposed, zero-padded to Npad columns.
// C  f32 (M x N). M % 64 == 0, K % 32 == 0, Npad % 128 == 0; gridDim.y == Npad/128.
#define BS_STRIDE 40   // u16 stride per column in LDS: 80B (16B aligned, bank-spread)
__global__ __launch_bounds__(256) void gemm_bf16_wmma(
    const u16* __restrict__ A, const u16* __restrict__ Bt,
    float* __restrict__ C, int M, int N, int K, float alpha, int accum) {
  __shared__ u16 Bs[128 * BS_STRIDE];
  const int tid  = threadIdx.x;
  const int wave = tid >> 5, lane = tid & 31;
  const int wm = wave >> 1, wn = wave & 1;          // 4 M-waves x 2 N-waves
  const int mBase  = blockIdx.x * 64 + wm * 16;
  const int nBlock = blockIdx.y * 128;
  const int mRow = lane & 15;
  const int kh   = lane >> 4;                        // half-lane group

  v8f acc[4] = {};

  const int stC    = tid >> 1;                       // 0..127: tile column
  const int stHalf = (tid & 1) * 16;                 // 0/16: k offset within tile

  for (int k0 = 0; k0 < K; k0 += 32) {
    __syncthreads();
    // stage B tile column-major: Bs[c][k], 32 k-values contiguous per column
    const u16* src = Bt + (size_t)(nBlock + stC) * K + k0 + stHalf;
    uint4 d0 = *reinterpret_cast<const uint4*>(src);
    uint4 d1 = *reinterpret_cast<const uint4*>(src + 8);
    u16* dst = &Bs[stC * BS_STRIDE + stHalf];
    *reinterpret_cast<uint4*>(dst)     = d0;
    *reinterpret_cast<uint4*>(dst + 8) = d1;
    __syncthreads();

    // A fragment: lane holds row mRow, K = {k0+kh*8..+7} U {k0+16+kh*8..+7}
    const u16* arow = A + (size_t)(mBase + mRow) * K + k0;
    if (k0 + 32 < K) __builtin_prefetch(arow + 32, 0, 3);
    union { uint4 q[2]; v16u u; } af;
    af.q[0] = *reinterpret_cast<const uint4*>(arow + kh * 8);
    af.q[1] = *reinterpret_cast<const uint4*>(arow + 16 + kh * 8);
    v16bf av = __builtin_bit_cast(v16bf, af.u);

    // load ALL B fragments, fence (ties accumulators so WMMAs can't hoist above),
    // then issue the 4 WMMAs back-to-back with a single dscnt wait.
    union BF { uint4 q[2]; v16u u; } bfr[4];
#pragma unroll
    for (int nt = 0; nt < 4; ++nt) {
      const int bcol = wn * 64 + nt * 16 + (lane & 15);
      const u16* bp = &Bs[bcol * BS_STRIDE + kh * 8];
      bfr[nt].q[0] = *reinterpret_cast<const uint4*>(bp);
      bfr[nt].q[1] = *reinterpret_cast<const uint4*>(bp + 16);
    }
    asm volatile("" : "+v"(acc[0]), "+v"(acc[1]), "+v"(acc[2]), "+v"(acc[3])
                 :: "memory");
#pragma unroll
    for (int nt = 0; nt < 4; ++nt) {
      v16bf bv = __builtin_bit_cast(v16bf, bfr[nt].u);
      acc[nt] = __builtin_amdgcn_wmma_f32_16x16x32_bf16(
          false, av, false, bv, (short)0, acc[nt], false, false);
    }
  }

#pragma unroll
  for (int nt = 0; nt < 4; ++nt) {
    const int col = nBlock + wn * 64 + nt * 16 + (lane & 15);
    if (col < N) {
#pragma unroll
      for (int r = 0; r < 8; ++r) {
        const int row = mBase + kh * 8 + r;
        size_t idx = (size_t)row * N + col;
        float v = alpha * acc[nt][r];
        C[idx] = accum ? (C[idx] + v) : v;
      }
    }
  }
}

// ---------------- rmsnorm (D elems/row) -> bf16, one wave per row ----------------
__global__ void rms_to_bf16_k(const float* __restrict__ in, const float* __restrict__ w,
                              u16* __restrict__ out, int D) {
  int wave = threadIdx.x >> 5, lane = threadIdx.x & 31;
  int row = blockIdx.x * 8 + wave;
  const float* x = in + (size_t)row * D;
  float s = 0.f;
  for (int j = lane; j < D; j += 32) { float v = x[j]; s += v * v; }
#pragma unroll
  for (int off = 16; off; off >>= 1) s += __shfl_xor(s, off, 32);
  float inv = rsqrtf(s / (float)D + EPS_);
  for (int j = lane; j < D; j += 32) out[(size_t)row * D + j] = f2bf(x[j] * inv * w[j]);
}

// ---------------- depthwise causal conv (token coords, dir-aware) + silu ----------------
__global__ void conv_silu_k(const float* __restrict__ zx, const float* __restrict__ cw,
                            const float* __restrict__ cb, float* __restrict__ xc, int dir) {
  int idx = blockIdx.x * blockDim.x + threadIdx.x;
  if (idx >= ROWS_ * CDIM_) return;
  int c = idx % CDIM_;
  int l = (idx / CDIM_) & 63;
  int b = idx / (CDIM_ * 64);
  float acc = cb[c];
#pragma unroll
  for (int k = 0; k < 4; ++k) {
    int ls = dir ? (l + 3 - k) : (l - 3 + k);
    if (ls >= 0 && ls < 64)
      acc += cw[c * 4 + k] * zx[(size_t)(b * 64 + ls) * DPROJ_ + DIN_ + c];
  }
  xc[idx] = siluf(acc);
}

// ---------------- dt = softplus(raw + dtb); decay = exp(-exp(Alog)*dt) ----------------
__global__ void dt_decay_k(const float* __restrict__ zx, const float* __restrict__ dtb,
                           const float* __restrict__ Alog, float* __restrict__ dt,
                           float* __restrict__ dec) {
  int idx = blockIdx.x * blockDim.x + threadIdx.x;
  if (idx >= ROWS_ * H_) return;
  int h = idx & (H_ - 1);
  float v = zx[(size_t)(idx >> 3) * DPROJ_ + DIN_ + CDIM_ + h] + dtb[h];
  float d = (v > 20.f) ? v : log1pf(expf(v));
  dt[idx]  = d;
  dec[idx] = expf(-expf(Alog[h]) * d);
}

// ---------------- selective scan: one block per (b,h); state (64p x 64n) in regs ----------------
__global__ __launch_bounds__(256) void scan_k(const float* __restrict__ xc,
                                              const float* __restrict__ dt,
                                              const float* __restrict__ dec,
                                              const float* __restrict__ Dp,
                                              float* __restrict__ y, int dir) {
  __shared__ float sB[NST_], sC[NST_], sX[DH_], sScal[2];
  int b = blockIdx.x >> 3, h = blockIdx.x & 7;
  int t = threadIdx.x;
  int p = t >> 2, q = t & 3;                 // thread owns (p, n = q*16..q*16+15)
  float st[16];
#pragma unroll
  for (int j = 0; j < 16; ++j) st[j] = 0.f;
  float dph = Dp[h];

  for (int s = 0; s < 64; ++s) {
    int l = dir ? (63 - s) : s;
    size_t rb = (size_t)(b * 64 + l);
    if (t < 64)        sB[t]       = xc[rb * CDIM_ + DIN_ + t];
    else if (t < 128)  sC[t - 64]  = xc[rb * CDIM_ + DIN_ + NST_ + (t - 64)];
    else if (t < 192)  sX[t - 128] = xc[rb * CDIM_ + h * DH_ + (t - 128)];
    else if (t == 192) { sScal[0] = dt[rb * H_ + h]; sScal[1] = dec[rb * H_ + h]; }
    __syncthreads();

    float xs = sX[p];
    float coeff = sScal[0] * xs;
    float dcy = sScal[1];
    float part = 0.f;
#pragma unroll
    for (int j = 0; j < 16; ++j) {
      int n = q * 16 + j;
      float sv = dcy * st[j] + coeff * sB[n];
      st[j] = sv;
      part += sv * sC[n];
    }
    part += __shfl_xor(part, 1, 32);
    part += __shfl_xor(part, 2, 32);
    if (q == 0) y[rb * DIN_ + h * DH_ + p] = part + dph * xs;
    __syncthreads();
  }
}

// ---------------- gated rmsnorm over DIN=512: out = rmsnorm(y*silu(z)) * gn -> bf16 ----------------
__global__ void gate_norm_k(const float* __restrict__ y, const float* __restrict__ zx,
                            const float* __restrict__ gn, u16* __restrict__ out) {
  int wave = threadIdx.x >> 5, lane = threadIdx.x & 31;
  int row = blockIdx.x * 8 + wave;
  const float* yr = y + (size_t)row * DIN_;
  const float* zr = zx + (size_t)row * DPROJ_;
  float g[16];
  float s = 0.f;
#pragma unroll
  for (int i = 0; i < 16; ++i) {
    int j = lane + i * 32;
    float gg = yr[j] * siluf(zr[j]);
    g[i] = gg;
    s += gg * gg;
  }
#pragma unroll
  for (int off = 16; off; off >>= 1) s += __shfl_xor(s, off, 32);
  float inv = rsqrtf(s / (float)DIN_ + EPS_);
#pragma unroll
  for (int i = 0; i < 16; ++i) {
    int j = lane + i * 32;
    out[(size_t)row * DIN_ + j] = f2bf(g[i] * inv * gn[j]);
  }
}

// ---------------- final norm inv, pooling, head ----------------
__global__ void row_inv_k(const float* __restrict__ t, float* __restrict__ inv) {
  int wave = threadIdx.x >> 5, lane = threadIdx.x & 31;
  int row = blockIdx.x * 8 + wave;
  const float* x = t + (size_t)row * D_;
  float s = 0.f;
  for (int j = lane; j < D_; j += 32) { float v = x[j]; s += v * v; }
#pragma unroll
  for (int off = 16; off; off >>= 1) s += __shfl_xor(s, off, 32);
  if (lane == 0) inv[row] = rsqrtf(s / (float)D_ + EPS_);
}

__global__ void pool_k(const float* __restrict__ t, const float* __restrict__ inv,
                       const float* __restrict__ fw, float* __restrict__ pooled) {
  int idx = blockIdx.x * blockDim.x + threadIdx.x;
  if (idx >= B_ * D_) return;
  int d = idx & (D_ - 1), b = idx >> 8;
  float s = 0.f;
  for (int l = 0; l < 64; ++l)
    s += t[((size_t)b * 64 + l) * D_ + d] * inv[b * 64 + l];
  pooled[idx] = s * fw[d] * (1.0f / 64.0f);
}

__global__ void head_k(const float* __restrict__ pooled, const float* __restrict__ hw,
                       const float* __restrict__ hb, float* __restrict__ out) {
  int idx = blockIdx.x * blockDim.x + threadIdx.x;
  if (idx >= B_ * NCLS_) return;
  int n = idx % NCLS_, b = idx / NCLS_;
  float s = hb[n];
  for (int d = 0; d < D_; ++d) s += pooled[b * D_ + d] * hw[d * NCLS_ + n];
  out[idx] = s;
}

// ---------------- host orchestration ----------------
static inline size_t alignup(size_t x) { return (x + 255) & ~(size_t)255; }

extern "C" void kernel_launch(void* const* d_in, const int* in_sizes, int n_in,
                              void* d_out, int out_size, void* d_ws, size_t ws_size,
                              hipStream_t stream) {
  (void)in_sizes; (void)n_in; (void)out_size; (void)ws_size;
  const float* x       = (const float*)d_in[0];
  const float* patch_w = (const float*)d_in[1];
  const float* patch_b = (const float*)d_in[2];
  const float* pos     = (const float*)d_in[3];
  const float* norms_w = (const float*)d_in[4];
  const float* final_w = (const float*)d_in[5];
  const float* head_w  = (const float*)d_in[6];
  const float* head_b  = (const float*)d_in[7];
  const float* Win_f  = (const float*)d_in[8];
  const float* cw_f   = (const float*)d_in[9];
  const float* cb_f   = (const float*)d_in[10];
  const float* dtb_f  = (const float*)d_in[11];
  const float* Alog_f = (const float*)d_in[12];
  const float* Dp_f   = (const float*)d_in[13];
  const float* gn_f   = (const float*)d_in[14];
  const float* Wout_f = (const float*)d_in[15];
  const float* Win_b  = (const float*)d_in[16];
  const float* cw_b   = (const float*)d_in[17];
  const float* cb_b   = (const float*)d_in[18];
  const float* dtb_b  = (const float*)d_in[19];
  const float* Alog_b = (const float*)d_in[20];
  const float* Dp_b   = (const float*)d_in[21];
  const float* gn_b   = (const float*)d_in[22];
  const float* Wout_b = (const float*)d_in[23];

  char* wp = (char*)d_ws;
  auto take = [&](size_t bytes) { void* p = (void*)wp; wp += alignup(bytes); return p; };
  float* t      = (float*)take((size_t)ROWS_ * D_ * 4);
  u16*   nxbf   = (u16*)  take((size_t)ROWS_ * D_ * 2);
  u16*   im2    = (u16*)  take((size_t)ROWS_ * 768 * 2);
  float* zx     = (float*)take((size_t)ROWS_ * DPROJ_ * 4);
  float* xc     = (float*)take((size_t)ROWS_ * CDIM_ * 4);
  float* dtbuf  = (float*)take((size_t)ROWS_ * H_ * 4);
  float* decbuf = (float*)take((size_t)ROWS_ * H_ * 4);
  float* ybuf   = (float*)take((size_t)ROWS_ * DIN_ * 4);
  u16*   ygbf   = (u16*)  take((size_t)ROWS_ * DIN_ * 2);
  u16*   pwbf   = (u16*)  take((size_t)768 * 256 * 2);        // N=256 x K=768 (= patch_w as-is)
  u16*   winfbf = (u16*)  take((size_t)2 * DPROJP * D_ * 2);  // per layer: Npad=1280 x K=256
  u16*   winbbf = (u16*)  take((size_t)2 * DPROJP * D_ * 2);
  u16*   woutfbf= (u16*)  take((size_t)2 * DIN_ * D_ * 2);    // per layer: N=256 x K=512
  u16*   woutbbf= (u16*)  take((size_t)2 * DIN_ * D_ * 2);
  float* rinv   = (float*)take((size_t)ROWS_ * 4);
  float* pooled = (float*)take((size_t)B_ * D_ * 4);

  // weight conversions (all GEMM B-operands stored pre-transposed: Npad x K, pad zeroed)
  {
    hipMemsetAsync(winfbf, 0, (size_t)2 * DPROJP * D_ * 2, stream);
    hipMemsetAsync(winbbf, 0, (size_t)2 * DPROJP * D_ * 2, stream);
    int n = 2 * D_ * DPROJ_;
    cvtT_bf16_k<<<(n + 255) / 256, 256, 0, stream>>>(Win_f, winfbf, 2, D_, DPROJ_, DPROJP);
    cvtT_bf16_k<<<(n + 255) / 256, 256, 0, stream>>>(Win_b, winbbf, 2, D_, DPROJ_, DPROJP);
    n = 2 * DIN_ * D_;
    cvtT_bf16_k<<<(n + 255) / 256, 256, 0, stream>>>(Wout_f, woutfbf, 2, DIN_, D_, D_);
    cvtT_bf16_k<<<(n + 255) / 256, 256, 0, stream>>>(Wout_b, woutbbf, 2, DIN_, D_, D_);
    n = 256 * 768;   // patch_w is (D, 768) = (N, K) already, N=256 needs no pad
    cvt_bf16_k<<<(n + 255) / 256, 256, 0, stream>>>(patch_w, pwbf, n);
  }

  // patch embed
  im2col_k<<<(ROWS_ * 768 + 255) / 256, 256, 0, stream>>>(x, im2);
  gemm_bf16_wmma<<<dim3(ROWS_ / 64, 256 / 128), 256, 0, stream>>>(
      im2, pwbf, t, ROWS_, D_, 768, 1.0f, 0);
  bias_pos_k<<<(ROWS_ * D_ + 255) / 256, 256, 0, stream>>>(t, patch_b, pos);

  for (int i = 0; i < 2; ++i) {
    rms_to_bf16_k<<<ROWS_ / 8, 256, 0, stream>>>(t, norms_w + i * D_, nxbf, D_);
    for (int dir = 0; dir < 2; ++dir) {
      const u16*   Winbf  = (dir ? winbbf : winfbf) + (size_t)i * DPROJP * D_;
      const u16*   Woutbf = (dir ? woutbbf : woutfbf) + (size_t)i * DIN_ * D_;
      const float* cw  = (dir ? cw_b  : cw_f)  + i * CDIM_ * 4;
      const float* cb  = (dir ? cb_b  : cb_f)  + i * CDIM_;
      const float* dtb = (dir ? dtb_b : dtb_f) + i * H_;
      const float* Al  = (dir ? Alog_b: Alog_f)+ i * H_;
      const float* Dpp = (dir ? Dp_b  : Dp_f)  + i * H_;
      const float* gnp = (dir ? gn_b  : gn_f)  + i * DIN_;

      gemm_bf16_wmma<<<dim3(ROWS_ / 64, DPROJP / 128), 256, 0, stream>>>(
          nxbf, Winbf, zx, ROWS_, DPROJ_, D_, 1.0f, 0);
      conv_silu_k<<<(ROWS_ * CDIM_ + 255) / 256, 256, 0, stream>>>(zx, cw, cb, xc, dir);
      dt_decay_k<<<(ROWS_ * H_ + 255) / 256, 256, 0, stream>>>(zx, dtb, Al, dtbuf, decbuf);
      scan_k<<<B_ * H_, 256, 0, stream>>>(xc, dtbuf, decbuf, Dpp, ybuf, dir);
      gate_norm_k<<<ROWS_ / 8, 256, 0, stream>>>(ybuf, zx, gnp, ygbf);
      gemm_bf16_wmma<<<dim3(ROWS_ / 64, 256 / 128), 256, 0, stream>>>(
          ygbf, Woutbf, t, ROWS_, D_, DIN_, 0.5f, 1);
    }
  }

  row_inv_k<<<ROWS_ / 8, 256, 0, stream>>>(t, rinv);
  pool_k<<<(B_ * D_ + 255) / 256, 256, 0, stream>>>(t, rinv, final_w, pooled);
  head_k<<<(B_ * NCLS_ + 255) / 256, 256, 0, stream>>>(pooled, head_w, head_b, (float*)d_out);
}
